// MixtralAttention_41686952575381
// MI455X (gfx1250) — matrix operand order repaired and checked by reading, not compiled, not run on hardware
//
#include <hip/hip_runtime.h>
#include <hip/hip_bf16.h>

#define S_LEN 2048
#define HID   4096
#define NH    32
#define NKV   8
#define HD    128
#define QSZ   4096
#define KVSZ  1024
#define QKV_N 6144
#define SOFT_SCALE 0.08838834764831845f  /* 128^-0.5 */

typedef __bf16 v8bf  __attribute__((ext_vector_type(8)));
typedef __bf16 v16bf __attribute__((ext_vector_type(16)));
typedef float  v8f   __attribute__((ext_vector_type(8)));

// ---- WMMA helpers -----------------------------------------------------------
__device__ __forceinline__ v8f wmma_bf16(v16bf a, v16bf b, v8f c) {
  return __builtin_amdgcn_wmma_f32_16x16x32_bf16(false, a, false, b, (short)0, c,
                                                 false, false);
}

// A-matrix 16x32 bf16 fragment: lane L (m=L%16, h=L/16) holds
// frag[0..7]  = A[m][k0 + 16h .. k0 + 16h + 7]
// frag[8..15] = A[m][k0 + 16h + 8 .. k0 + 16h + 15]
__device__ __forceinline__ v16bf ld_a_frag(const __bf16* rowp, int k0, int h) {
  v8bf lo = *(const v8bf*)(rowp + k0 + 16 * h);
  v8bf hi = *(const v8bf*)(rowp + k0 + 16 * h + 8);
  return __builtin_shufflevector(lo, hi, 0, 1, 2, 3, 4, 5, 6, 7, 8, 9, 10, 11,
                                 12, 13, 14, 15);
}

// B-matrix 32x16 bf16 fragment from Bt[N][K] (row-major, K contiguous):
// lane L (n=L%16, h=L/16) holds frag[e] = B[K = k0 + 16h + e][n]  -> one 32B load
__device__ __forceinline__ v16bf ld_b_frag(const __bf16* btrow, int k0, int h) {
  return *(const v16bf*)(btrow + k0 + 16 * h);
}

// ---- CDNA5 async global->LDS (ASYNCcnt-tracked, no VGPR round trip) ---------
// LDS byte offset = low 32 bits of the flat shared pointer (LDS aperture is in
// the high 32 bits on gfx1250).
__device__ __forceinline__ void async_g2l_b128(void* lds_dst, const void* gsrc) {
  unsigned loff = (unsigned)(unsigned long long)lds_dst;
  asm volatile("global_load_async_to_lds_b128 %0, %1, off"
               :
               : "v"(loff), "v"(gsrc)
               : "memory");
}
__device__ __forceinline__ void wait_async0() {
#if __has_builtin(__builtin_amdgcn_s_wait_asynccnt)
  __builtin_amdgcn_s_wait_asynccnt(0);
#else
  asm volatile("s_wait_asynccnt 0x0" ::: "memory");
#endif
}

// ---- precision/layout prep kernels -----------------------------------------
__global__ void k_f32_to_bf16(const float* __restrict__ in,
                              __bf16* __restrict__ out, long n) {
  long i = (long)blockIdx.x * blockDim.x + threadIdx.x;
  if (i < n) out[i] = (__bf16)in[i];
}

// in: fp32 [R][C] row-major -> out: bf16 [C][R] (i.e. Bt layout)
__global__ void k_transpose_f32_to_bf16(const float* __restrict__ in,
                                        __bf16* __restrict__ out, int R, int C) {
  long i = (long)blockIdx.x * blockDim.x + threadIdx.x;
  long n = (long)R * C;
  if (i >= n) return;
  int r = (int)(i / C);
  int c = (int)(i % C);
  out[(long)c * R + r] = (__bf16)in[i];
}

// ---- bf16 WMMA GEMM with async-LDS B staging --------------------------------
// C[M][N] (fp32) = A[M][K] * Bt[N][K]^T
// Block = 128 threads (4 waves) computing a 64(M) x 64(N) tile.
// All 4 waves share one B tile [64 x 32] staged in LDS via
// GLOBAL_LOAD_ASYNC_TO_LDS_B128, double-buffered across k-steps.
#define BPAD 48  /* row pitch in bf16 elems: 96 B -> 32B-aligned v16bf reads */
__global__ __launch_bounds__(128) void k_gemm_bf16(const __bf16* __restrict__ A,
                                                   const __bf16* __restrict__ Bt,
                                                   float* __restrict__ C, int M,
                                                   int N, int K) {
  const int t = threadIdx.x;        // 0..127
  const int wave = t >> 5;          // 0..3 -> 16-row slice of the 64-row tile
  const int lane = t & 31;
  const int h = lane >> 4;
  const int nl = lane & 15;
  const int mt = blockIdx.x;        // 64-row tile
  const int nt = blockIdx.y;        // 64-col tile

  __shared__ __align__(32) __bf16 bbuf[2][64][BPAD];

  const __bf16* arow = A + (long)(mt * 64 + wave * 16 + nl) * K;
  const __bf16* bbase = Bt + (long)(nt * 64) * K;

  // stage one [64 rows x 32 k] B slice: 256 16-byte chunks over 128 threads
  auto stage = [&](int buf, int k0) {
#pragma unroll
    for (int i = 0; i < 2; ++i) {
      int c = t + i * 128;
      int row = c >> 2;
      int sub = c & 3;  // 16B sub-chunk within the 64B row slice
      async_g2l_b128(&bbuf[buf][row][sub * 8],
                     bbase + (long)row * K + k0 + sub * 8);
    }
  };

  v8f acc0 = {}, acc1 = {}, acc2 = {}, acc3 = {};
  stage(0, 0);
  for (int k0 = 0; k0 < K; k0 += 32) {
    const int buf = (k0 >> 5) & 1;
    wait_async0();
    __syncthreads();
    if (k0 + 32 < K) stage(buf ^ 1, k0 + 32);  // overlap DMA with WMMA

    v16bf a = ld_a_frag(arow, k0, h);
    v16bf b0 = *(const v16bf*)(&bbuf[buf][0 * 16 + nl][16 * h]);
    v16bf b1 = *(const v16bf*)(&bbuf[buf][1 * 16 + nl][16 * h]);
    v16bf b2 = *(const v16bf*)(&bbuf[buf][2 * 16 + nl][16 * h]);
    v16bf b3 = *(const v16bf*)(&bbuf[buf][3 * 16 + nl][16 * h]);
    acc0 = wmma_bf16(a, b0, acc0);
    acc1 = wmma_bf16(a, b1, acc1);
    acc2 = wmma_bf16(a, b2, acc2);
    acc3 = wmma_bf16(a, b3, acc3);
  }
  // C/D layout: lane holds col n = lane%16 (within 16-chunk), VGPR r -> row r+8h
#pragma unroll
  for (int r = 0; r < 8; ++r) {
    long row = mt * 64 + wave * 16 + r + 8 * h;
    long base = row * N + nt * 64 + nl;
    C[base + 0] = acc0[r];
    C[base + 16] = acc1[r];
    C[base + 32] = acc2[r];
    C[base + 48] = acc3[r];
  }
}

// ---- RoPE + scatter to attention layouts -----------------------------------
// qkv fp32 [S][6144] -> Qr bf16 [S][4096] (roped),
//                        Kr bf16 [kv][S][128] (roped; Bt layout for Q*K^T),
//                        Vt bf16 [kv][128][S]  (Bt layout for P*V)
__global__ void k_rope_scatter(const float* __restrict__ qkv,
                               const int* __restrict__ pos,
                               __bf16* __restrict__ Qr, __bf16* __restrict__ Kr,
                               __bf16* __restrict__ Vt) {
  long idx = (long)blockIdx.x * blockDim.x + threadIdx.x;
  long total = (long)S_LEN * QKV_N;
  if (idx >= total) return;
  int s = (int)(idx / QKV_N);
  int c = (int)(idx % QKV_N);
  float v = qkv[idx];
  if (c < QSZ + KVSZ) {
    int d = c % HD;
    int i = d & 63;
    float p = (float)pos[s];
    float freq = powf(10000.0f, -(float)i * (1.0f / 64.0f));
    float ang = p * freq;
    float cs = cosf(ang), sn = sinf(ang);
    int pc = (d < 64) ? c + 64 : c - 64;
    float pv = qkv[(long)s * QKV_N + pc];
    float o = (d < 64) ? (v * cs - pv * sn) : (v * cs + pv * sn);
    if (c < QSZ) {
      Qr[(long)s * QSZ + c] = (__bf16)o;
    } else {
      int cc = c - QSZ;
      int kv = cc / HD;
      Kr[((long)kv * S_LEN + s) * HD + d] = (__bf16)o;
    }
  } else {
    int cc = c - QSZ - KVSZ;
    int kv = cc / HD;
    int d = cc % HD;
    Vt[((long)kv * HD + d) * S_LEN + s] = (__bf16)v;
  }
}

// ---- flash attention: 1 wave per (16-query block, head) --------------------
__global__ __launch_bounds__(32) void k_flash_attn(const __bf16* __restrict__ Qr,
                                                   const __bf16* __restrict__ Kr,
                                                   const __bf16* __restrict__ Vt,
                                                   __bf16* __restrict__ O) {
  const int lane = threadIdx.x;
  const int h = lane >> 4;
  const int nl = lane & 15;
  const int qb = blockIdx.x;   // 0..127 (16 queries each)
  const int head = blockIdx.y; // 0..31
  const int kv = head >> 2;    // GQA: 4 q-heads per kv head

  const __bf16* qrow = Qr + (long)(qb * 16 + nl) * QSZ + head * HD;
  const __bf16* Kh = Kr + (long)kv * S_LEN * HD;
  const __bf16* Vh = Vt + (long)kv * HD * S_LEN;

  __shared__ __align__(32) __bf16 plds[16][32];

  v8f acc[8] = {};
  float mrow[8], lrow[8];
#pragma unroll
  for (int r = 0; r < 8; ++r) {
    mrow[r] = -3e38f;
    lrow[r] = 0.0f;
  }

  const int qhi = qb * 16 + 15;
  const int nkb = qhi / 32 + 1;

  for (int jb = 0; jb < nkb; ++jb) {
    const int j = jb * 32;
    // ---- scores: 16 queries x 32 keys, K-dim = 128 ----
    v8f c0 = {}, c1 = {};
    const __bf16* krow0 = Kh + (long)(j + nl) * HD;
    const __bf16* krow1 = Kh + (long)(j + 16 + nl) * HD;
#pragma unroll
    for (int k0 = 0; k0 < HD; k0 += 32) {
      v16bf aq = ld_a_frag(qrow, k0, h);
      c0 = wmma_bf16(aq, ld_b_frag(krow0, k0, h), c0);
      c1 = wmma_bf16(aq, ld_b_frag(krow1, k0, h), c1);
    }
    // ---- online softmax (rows live in 16-lane halves) ----
    float alpha[8];
#pragma unroll
    for (int r = 0; r < 8; ++r) {
      int q = qb * 16 + r + 8 * h;
      float s0 = c0[r] * SOFT_SCALE;
      float s1 = c1[r] * SOFT_SCALE;
      if (j + nl > q) s0 = -3e38f;
      if (j + 16 + nl > q) s1 = -3e38f;
      float mx = fmaxf(s0, s1);
#pragma unroll
      for (int msk = 1; msk < 16; msk <<= 1)
        mx = fmaxf(mx, __shfl_xor(mx, msk, 16));
      float mnew = fmaxf(mrow[r], mx);
      alpha[r] = __expf(mrow[r] - mnew);
      float e0 = __expf(s0 - mnew);
      float e1 = __expf(s1 - mnew);
      float rs = e0 + e1;
#pragma unroll
      for (int msk = 1; msk < 16; msk <<= 1) rs += __shfl_xor(rs, msk, 16);
      lrow[r] = lrow[r] * alpha[r] + rs;
      mrow[r] = mnew;
      plds[r + 8 * h][nl] = (__bf16)e0;
      plds[r + 8 * h][16 + nl] = (__bf16)e1;
    }
#pragma unroll
    for (int d = 0; d < 8; ++d)
#pragma unroll
      for (int r = 0; r < 8; ++r) acc[d][r] *= alpha[r];

    __syncthreads();
    // P as A-fragment (C-layout -> A-layout reshape through LDS)
    v8bf plo = *(const v8bf*)(&plds[nl][16 * h]);
    v8bf phi = *(const v8bf*)(&plds[nl][16 * h + 8]);
    v16bf ap = __builtin_shufflevector(plo, phi, 0, 1, 2, 3, 4, 5, 6, 7, 8, 9,
                                       10, 11, 12, 13, 14, 15);
    // ---- P * V : 8 d-chunks of 16 ----
#pragma unroll
    for (int d = 0; d < 8; ++d) {
      const __bf16* vrow = Vh + (long)(d * 16 + nl) * S_LEN;
      acc[d] = wmma_bf16(ap, ld_b_frag(vrow, j, h), acc[d]);
    }
    __syncthreads();
  }

  // ---- epilogue: O[q][head*128 + d] = acc / l ----
#pragma unroll
  for (int d = 0; d < 8; ++d) {
#pragma unroll
    for (int r = 0; r < 8; ++r) {
      long row = qb * 16 + r + 8 * h;
      long col = head * HD + d * 16 + nl;
      O[row * QSZ + col] = (__bf16)(acc[d][r] / lrow[r]);
    }
  }
}

// ---- launcher ---------------------------------------------------------------
extern "C" void kernel_launch(void* const* d_in, const int* in_sizes, int n_in,
                              void* d_out, int out_size, void* d_ws,
                              size_t ws_size, hipStream_t stream) {
  const int* positions = (const int*)d_in[0];
  const float* hs = (const float*)d_in[1];
  const float* wqkv = (const float*)d_in[2];
  const float* wo = (const float*)d_in[3];
  float* out = (float*)d_out;

  char* ws = (char*)d_ws;
  size_t off = 0;
  __bf16* hsb = (__bf16*)(ws + off);   off += (size_t)S_LEN * HID * 2;   // 16 MB
  __bf16* wqkvb = (__bf16*)(ws + off); off += (size_t)QKV_N * HID * 2;   // 48 MB
  __bf16* wob = (__bf16*)(ws + off);   off += (size_t)HID * QSZ * 2;     // 32 MB
  size_t qkv_off = off;
  float* qkvf = (float*)(ws + off);    off += (size_t)S_LEN * QKV_N * 4; // 50 MB
  __bf16* Qr = (__bf16*)(ws + off);    off += (size_t)S_LEN * QSZ * 2;   // 16 MB
  __bf16* Kr = (__bf16*)(ws + off);    off += (size_t)NKV * S_LEN * HD * 2;
  __bf16* Vt = (__bf16*)(ws + off);    off += (size_t)NKV * S_LEN * HD * 2;
  __bf16* attnb = (__bf16*)(ws + qkv_off); // reuse qkv fp32 region (dead by then)

  // 1) hidden fp32 -> bf16
  {
    long n = (long)S_LEN * HID;
    k_f32_to_bf16<<<dim3((unsigned)((n + 255) / 256)), dim3(256), 0, stream>>>(
        hs, hsb, n);
  }
  // 2) w_qkv [4096][6144] -> Bt [6144][4096] bf16
  {
    long n = (long)HID * QKV_N;
    k_transpose_f32_to_bf16<<<dim3((unsigned)((n + 255) / 256)), dim3(256), 0,
                              stream>>>(wqkv, wqkvb, HID, QKV_N);
  }
  // 3) w_o [4096][4096] -> Bt [HID][QSZ] bf16
  {
    long n = (long)QSZ * HID;
    k_transpose_f32_to_bf16<<<dim3((unsigned)((n + 255) / 256)), dim3(256), 0,
                              stream>>>(wo, wob, QSZ, HID);
  }
  // 4) QKV GEMM: [2048][4096] x [4096][6144] -> fp32 [2048][6144]
  k_gemm_bf16<<<dim3(S_LEN / 64, QKV_N / 64), dim3(128), 0, stream>>>(
      hsb, wqkvb, qkvf, S_LEN, QKV_N, HID);
  // 5) RoPE + scatter to attention layouts
  {
    long n = (long)S_LEN * QKV_N;
    k_rope_scatter<<<dim3((unsigned)((n + 255) / 256)), dim3(256), 0, stream>>>(
        qkvf, positions, Qr, Kr, Vt);
  }
  // 6) causal GQA flash attention -> attnb bf16 [2048][4096]
  k_flash_attn<<<dim3(S_LEN / 16, NH), dim3(32), 0, stream>>>(Qr, Kr, Vt, attnb);
  // 7) O-projection: [2048][4096] x [4096][4096] -> d_out fp32
  k_gemm_bf16<<<dim3(S_LEN / 64, HID / 64), dim3(128), 0, stream>>>(
      attnb, wob, out, S_LEN, HID, QSZ);
}
